// PointNet2_79688823210374
// MI455X (gfx1250) — matrix operand order repaired and checked by reading, not compiled, not run on hardware
//
#include <hip/hip_runtime.h>
#include <hip/hip_bf16.h>
#include <cstdint>

typedef __attribute__((ext_vector_type(16))) _Float16 v16h;
typedef __attribute__((ext_vector_type(8)))  _Float16 h8;
typedef __attribute__((ext_vector_type(8)))  float    v8f;

#define BM 128
#define BN 64
#define BK 32
#define LDA (BK + 8)   // padded halves per LDS row (80B stride, 16B aligned, conflict-free)

// CDNA5 async global->LDS copy (ASYNCcnt-tracked), 16 bytes per lane.
#define ASYNC_B128(ldsoff, gptr)                                               \
  asm volatile("global_load_async_to_lds_b128 %0, %1, off"                     \
               :: "v"(ldsoff), "v"((unsigned long long)(uintptr_t)(gptr))      \
               : "memory")

#if __has_builtin(__builtin_amdgcn_s_wait_asynccnt)
#define WAIT_ASYNC(n) __builtin_amdgcn_s_wait_asynccnt(n)
#else
#define WAIT_ASYNC(n) asm volatile("s_wait_asynccnt %0" :: "i"(n) : "memory")
#endif

// ---------------------------------------------------------------------------
// WMMA GEMM: Y[M,Co] = Xh[M,Kp](f16) * Wh[Cop,Kp](f16)^T + bias, f32 accum.
// Kp multiple of 32 (zero-padded), M multiple of 128, Cop multiple of 64.
// Per-channel sum/sumsq accumulated into ssum/ssq (fused BN statistics).
// Double-buffered LDS filled by async b128 copies, pipelined against WMMA.
// ---------------------------------------------------------------------------
__global__ __launch_bounds__(256) void k_gemm_wmma(
    const _Float16* __restrict__ Xh, const _Float16* __restrict__ Wh,
    const float* __restrict__ bias, float* __restrict__ Y,
    float* __restrict__ ssum, float* __restrict__ ssq,
    int M, int Kp, int Co)
{
  __shared__ __align__(16) _Float16 As[2][BM][LDA];
  __shared__ __align__(16) _Float16 BsT[2][BN][LDA];

  const int tid  = threadIdx.x;
  const int lane = tid & 31;
  const int wave = tid >> 5;       // 8 waves
  const int wm   = wave >> 2;      // 0..1 : 64-row slab
  const int wn   = wave & 3;       // 0..3 : 16-col slab
  const int bm0  = blockIdx.x * BM;
  const int bn0  = blockIdx.y * BN;

  // staging geometry: each thread moves three 16B chunks per K-tile
  const int ar = tid >> 2;             // 0..63 (row)
  const int ac = (tid & 3) * 8;        // half-column 0,8,16,24
  const _Float16* gA0 = Xh + (size_t)(bm0 + ar) * Kp + ac;
  const _Float16* gA1 = gA0 + (size_t)64 * Kp;
  const _Float16* gB  = Wh + (size_t)(bn0 + ar) * Kp + ac;
  uint32_t dA0[2], dA1[2], dB[2];
#pragma unroll
  for (int i = 0; i < 2; ++i) {
    dA0[i] = (uint32_t)(uintptr_t)&As[i][ar][ac];
    dA1[i] = (uint32_t)(uintptr_t)&As[i][64 + ar][ac];
    dB[i]  = (uint32_t)(uintptr_t)&BsT[i][ar][ac];
  }

  v8f acc[4] = {};
  const int nkt = Kp >> 5;

  // prime buffer 0 with K-tile 0
  ASYNC_B128(dA0[0], gA0);
  ASYNC_B128(dA1[0], gA1);
  ASYNC_B128(dB[0],  gB);

  for (int kt = 0; kt < nkt; ++kt) {
    __syncthreads();                 // all waves done reading buf[(kt+1)&1]
    if (kt + 1 < nkt) {              // uniform branch: EXEC untouched
      const int nb = (kt + 1) & 1;
      const int ko = (kt + 1) * 32;
      ASYNC_B128(dA0[nb], gA0 + ko);
      ASYNC_B128(dA1[nb], gA1 + ko);
      ASYNC_B128(dB[nb],  gB  + ko);
      WAIT_ASYNC(3);                 // tile kt (issued earlier) has landed
    } else {
      WAIT_ASYNC(0);
    }
    __syncthreads();                 // tile kt visible to every wave
    const int cb = kt & 1;

    const int half = lane >> 4;      // K-half select (ISA 16-bit A/B layouts)
    const int mn   = lane & 15;

    // B fragment: lane holds K = {half*8..+7} and {16+half*8..+7} of column (bn0+wn*16+mn)
    const h8* bp = (const h8*)(&BsT[cb][wn * 16 + mn][0]);
    h8 blo = bp[half];
    h8 bhi = bp[2 + half];
    v16h bf = __builtin_shufflevector(blo, bhi, 0,1,2,3,4,5,6,7,8,9,10,11,12,13,14,15);

#pragma unroll
    for (int t = 0; t < 4; ++t) {
      // A fragment: lane holds K = {half*8..+7} and {16+half*8..+7} of its row
      const h8* ap = (const h8*)(&As[cb][wm * 64 + t * 16 + mn][0]);
      h8 alo = ap[half];
      h8 ahi = ap[2 + half];
      v16h af = __builtin_shufflevector(alo, ahi, 0,1,2,3,4,5,6,7,8,9,10,11,12,13,14,15);
      acc[t] = __builtin_amdgcn_wmma_f32_16x16x32_f16(
          false, af, false, bf, (short)0, acc[t], false, false);
    }
  }

  // epilogue: bias + store f32 + fused per-channel BN statistics
  const int n  = lane & 15;
  const int mo = (lane >> 4) * 8;
  const int gn = bn0 + wn * 16 + n;
  if (gn < Co) {
    float bv = bias[gn];
    float ps = 0.f, pq = 0.f;
#pragma unroll
    for (int t = 0; t < 4; ++t) {
      int gmb = bm0 + wm * 64 + t * 16 + mo;
#pragma unroll
      for (int r = 0; r < 8; ++r) {
        float v = acc[t][r] + bv;
        Y[(size_t)(gmb + r) * Co + gn] = v;
        ps += v; pq += v * v;
      }
    }
    atomicAdd(&ssum[gn], ps);
    atomicAdd(&ssq[gn], pq);
  }
}

// pack W[Co,Ci] f32 -> Wh[Cop,Kp] f16, zero-padded
__global__ void k_prepack_w(const float* __restrict__ W, _Float16* __restrict__ Wh,
                            int Ci, int Co, int Kp, int Cop)
{
  int total = Cop * Kp;
  for (int e = blockIdx.x * 256 + threadIdx.x; e < total; e += gridDim.x * 256) {
    int o = e / Kp, c = e - o * Kp;
    float v = (o < Co && c < Ci) ? W[(size_t)o * Ci + c] : 0.f;
    Wh[e] = (_Float16)v;
  }
}

// ---------------------------------------------------------------------------
// Farthest point sampling: one 1024-thread block per batch (serial chain).
// ---------------------------------------------------------------------------
__global__ __launch_bounds__(1024) void k_fps(
    const float* __restrict__ xyz, int N, int bstride,
    float* __restrict__ dist, int* __restrict__ fidx, int S)
{
  const int b = blockIdx.x;
  const int tid = threadIdx.x;
  const float* px = xyz + (size_t)b * bstride;
  float* pd = dist + (size_t)b * N;
  __shared__ float sval[1024];
  __shared__ int   sidx[1024];
  for (int i = tid; i < N; i += 1024) pd[i] = 1e10f;
  int far = 0;
  __syncthreads();
  for (int s = 0; s < S; ++s) {
    if (tid == 0) fidx[(size_t)b * S + s] = far;
    float cx = px[far], cy = px[N + far], cz = px[2 * N + far];
    float best = -1.f; int bi = 0;
    for (int i = tid; i < N; i += 1024) {
      float dx = px[i] - cx, dy = px[N + i] - cy, dz = px[2 * N + i] - cz;
      float d = dx * dx + dy * dy + dz * dz;
      float dd = fminf(pd[i], d);
      pd[i] = dd;
      if (dd > best) { best = dd; bi = i; }
    }
    sval[tid] = best; sidx[tid] = bi;
    __syncthreads();
    for (int off = 512; off > 0; off >>= 1) {
      if (tid < off) {
        float ov = sval[tid + off]; int oi = sidx[tid + off];
        if (ov > sval[tid] || (ov == sval[tid] && oi < sidx[tid])) {
          sval[tid] = ov; sidx[tid] = oi;
        }
      }
      __syncthreads();
    }
    far = sidx[0];
    __syncthreads();
  }
}

__global__ void k_gather_xyz(const float* __restrict__ xyz, int N, int bstride,
                             const int* __restrict__ fidx,
                             float* __restrict__ nxyz, int B, int S)
{
  int t = blockIdx.x * 256 + threadIdx.x;
  if (t >= B * S) return;
  int b = t / S, s = t - b * S;
  int i = fidx[t];
  const float* px = xyz + (size_t)b * bstride;
  float* o = nxyz + (size_t)b * 3 * S;
  o[s] = px[i]; o[S + s] = px[N + i]; o[2 * S + s] = px[2 * N + i];
}

// ball query: first 32 indices (ascending) within radius; pad with first hit
__global__ void k_ball(const float* __restrict__ xyz, int N, int bstride,
                       const float* __restrict__ nxyz,
                       int* __restrict__ gidx, int B, int S, float r2)
{
  int t = blockIdx.x * 256 + threadIdx.x;
  if (t >= B * S) return;
  int b = t / S, s = t - b * S;
  const float* px = xyz + (size_t)b * bstride;
  const float* c  = nxyz + (size_t)b * 3 * S;
  float cx = c[s], cy = c[S + s], cz = c[2 * S + s];
  int* g = gidx + (size_t)t * 32;
  int cnt = 0, first = 0;
  for (int i = 0; i < N && cnt < 32; ++i) {
    float dx = px[i] - cx, dy = px[N + i] - cy, dz = px[2 * N + i] - cz;
    if (dx * dx + dy * dy + dz * dz <= r2) {
      if (cnt == 0) first = i;
      g[cnt++] = i;
    }
  }
  for (; cnt < 32; ++cnt) g[cnt] = first;
}

// grouped rows -> padded f16: G[(b,s,k), 0:3]=xyz-center, [3:3+C]=pts, pads=0
__global__ void k_group(const float* __restrict__ xyz, int N, int bstride,
                        const float* __restrict__ pts, int C,
                        const float* __restrict__ nxyz, const int* __restrict__ gidx,
                        _Float16* __restrict__ G, int B, int S, int Cg, int Kp)
{
  int total = B * S * 32 * Kp;
  for (int e = blockIdx.x * 256 + threadIdx.x; e < total; e += gridDim.x * 256) {
    int row = e / Kp, c = e - row * Kp;
    float v = 0.f;
    if (c < Cg) {
      int b = row / (S * 32);
      int rem = row - b * S * 32;
      int s = rem >> 5;
      int i = gidx[row];
      if (c < 3)
        v = xyz[(size_t)b * bstride + (size_t)c * N + i] -
            nxyz[(size_t)b * 3 * S + (size_t)c * S + s];
      else
        v = pts[(size_t)b * C * N + (size_t)(c - 3) * N + i];
    }
    G[e] = (_Float16)v;
  }
}

__global__ void k_zero_stats(float* s, float* q, int Co)
{
  int t = blockIdx.x * 256 + threadIdx.x;
  if (t < Co) { s[t] = 0.f; q[t] = 0.f; }
}

// BN+ReLU from fused stats; writes f32 in place and (optionally) next f16 input
__global__ void k_bnrelu(float* __restrict__ Y, _Float16* __restrict__ Xn,
                         int total, int Co, const float* __restrict__ s,
                         const float* __restrict__ q, const float* __restrict__ g,
                         const float* __restrict__ bt, float invM)
{
  for (int e = blockIdx.x * 256 + threadIdx.x; e < total; e += gridDim.x * 256) {
    int c = e % Co;
    float m = s[c] * invM;
    float var = fmaxf(q[c] * invM - m * m, 0.f);
    float v = fmaxf(g[c] * (Y[e] - m) * rsqrtf(var + 1e-5f) + bt[c], 0.f);
    Y[e] = v;
    if (Xn) Xn[e] = (_Float16)v;   // next-layer Kp == Co (Co multiple of 32)
  }
}

// max over the 32 grouped samples -> channel-major [B,Co,S]
__global__ void k_maxpool(const float* __restrict__ Y, float* __restrict__ out,
                          int B, int S, int Co)
{
  int t = blockIdx.x * 256 + threadIdx.x;
  if (t >= B * S * Co) return;
  int b = t / (S * Co);
  int rem = t - b * S * Co;
  int o = rem / S, s = rem - o * S;
  const float* y = Y + ((size_t)(b * S + s) * 32) * Co + o;
  float m = y[0];
  for (int k = 1; k < 32; ++k) m = fmaxf(m, y[(size_t)k * Co]);
  out[t] = m;                    // t == b*Co*S + o*S + s
}

// 3-NN + inverse-distance weights
__global__ void k_knn3(const float* __restrict__ x1, int N1, int b1s,
                       const float* __restrict__ x2, int N2, int b2s,
                       int* __restrict__ idx3, float* __restrict__ w3, int B)
{
  int t = blockIdx.x * 256 + threadIdx.x;
  if (t >= B * N1) return;
  int b = t / N1, n = t - b * N1;
  const float* p1 = x1 + (size_t)b * b1s;
  const float* p2 = x2 + (size_t)b * b2s;
  float cx = p1[n], cy = p1[N1 + n], cz = p1[2 * N1 + n];
  float d0 = 1e30f, d1 = 1e30f, d2 = 1e30f;
  int i0 = 0, i1 = 0, i2 = 0;
  for (int j = 0; j < N2; ++j) {
    float dx = p2[j] - cx, dy = p2[N2 + j] - cy, dz = p2[2 * N2 + j] - cz;
    float d = dx * dx + dy * dy + dz * dz;
    if (d < d0)      { d2 = d1; i2 = i1; d1 = d0; i1 = i0; d0 = d; i0 = j; }
    else if (d < d1) { d2 = d1; i2 = i1; d1 = d; i1 = j; }
    else if (d < d2) { d2 = d; i2 = j; }
  }
  float w0 = 1.f / (d0 + 1e-8f), w1 = 1.f / (d1 + 1e-8f), w2 = 1.f / (d2 + 1e-8f);
  float ws = w0 + w1 + w2;
  idx3[(size_t)t * 3]     = i0;  w3[(size_t)t * 3]     = w0 / ws;
  idx3[(size_t)t * 3 + 1] = i1;  w3[(size_t)t * 3 + 1] = w1 / ws;
  idx3[(size_t)t * 3 + 2] = i2;  w3[(size_t)t * 3 + 2] = w2 / ws;
}

// rows X[(b,n), 0:C1]=pts1, [C1:C1+C2]=interp(pts2) -> padded f16
__global__ void k_interp(const float* __restrict__ pts1, int C1,
                         const float* __restrict__ pts2, int C2,
                         int N1, int N2, const int* __restrict__ idx3,
                         const float* __restrict__ w3, _Float16* __restrict__ X,
                         int B, int Kp)
{
  int Ct = C1 + C2;
  int total = B * N1 * Kp;
  for (int e = blockIdx.x * 256 + threadIdx.x; e < total; e += gridDim.x * 256) {
    int row = e / Kp, c = e - row * Kp;
    float v = 0.f;
    if (c < Ct) {
      int b = row / N1, n = row - b * N1;
      if (c < C1) {
        v = pts1[(size_t)b * C1 * N1 + (size_t)c * N1 + n];
      } else {
        int cc = c - C1;
        const float* p2 = pts2 + (size_t)b * C2 * N2 + (size_t)cc * N2;
        const int*   id = idx3 + (size_t)row * 3;
        const float* w  = w3   + (size_t)row * 3;
        v = w[0] * p2[id[0]] + w[1] * p2[id[1]] + w[2] * p2[id[2]];
      }
    }
    X[e] = (_Float16)v;
  }
}

// rows [B*N, C] f32 -> channel-major [B, C, N]
__global__ void k_rows2chw(const float* __restrict__ Xr, float* __restrict__ out,
                           int B, int N, int C)
{
  int total = B * N * C;
  for (int e = blockIdx.x * 256 + threadIdx.x; e < total; e += gridDim.x * 256) {
    int row = e / C, c = e - row * C;
    int b = row / N, n = row - b * N;
    out[(size_t)b * C * N + (size_t)c * N + n] = Xr[e];
  }
}

// ---------------------------------------------------------------------------
static inline int cdiv(int a, int b) { return (a + b - 1) / b; }
static inline int r32(int x) { return (x + 31) & ~31; }
static inline int r64(int x) { return (x + 63) & ~63; }
static inline int capgrid(int total) {
  int g = cdiv(total, 256);
  return g > 8192 ? 8192 : g;
}

extern "C" void kernel_launch(void* const* d_in, const int* in_sizes, int n_in,
                              void* d_out, int out_size, void* d_ws, size_t ws_size,
                              hipStream_t stream)
{
  (void)in_sizes; (void)n_in; (void)out_size; (void)ws_size;
  const int B = 8, N0 = 4096;
  const float* xyz = (const float*)d_in[0];

  struct Lyr { const float *W, *b, *g, *bt; int ci, co; };
  int p = 1;
  auto take = [&](int ci, int co) {
    Lyr l; l.W = (const float*)d_in[p]; l.b = (const float*)d_in[p + 1];
    l.g = (const float*)d_in[p + 2]; l.bt = (const float*)d_in[p + 3];
    p += 4; l.ci = ci; l.co = co; return l;
  };
  Lyr sa1[3]; sa1[0]=take(9,32);   sa1[1]=take(32,32);   sa1[2]=take(32,64);
  Lyr sa2[3]; sa2[0]=take(67,64);  sa2[1]=take(64,64);   sa2[2]=take(64,128);
  Lyr sa3[3]; sa3[0]=take(131,128);sa3[1]=take(128,128); sa3[2]=take(128,256);
  Lyr sa4[3]; sa4[0]=take(259,256);sa4[1]=take(256,256); sa4[2]=take(256,512);
  Lyr fp4[2]; fp4[0]=take(768,256);fp4[1]=take(256,256);
  Lyr fp3[2]; fp3[0]=take(384,256);fp3[1]=take(256,256);
  Lyr fp2[2]; fp2[0]=take(320,256);fp2[1]=take(256,128);
  Lyr fp1[3]; fp1[0]=take(128,128);fp1[1]=take(128,128); fp1[2]=take(128,128);

  // ---- workspace bump allocator (bytes) ----
  char* base = (char*)d_ws;
  size_t off = 0;
  auto alloc = [&](size_t bytes) -> void* {
    off = (off + 255) & ~(size_t)255;
    void* ptr = base + off;
    off += bytes;
    return ptr;
  };
  float*    bufY  = (float*)   alloc((size_t)16777216 * 4); // f32 GEMM out (max 262144x64)
  _Float16* bufH0 = (_Float16*)alloc((size_t)8388608 * 2);  // f16 padded activations
  _Float16* bufH1 = (_Float16*)alloc((size_t)8388608 * 2);
  _Float16* Wh    = (_Float16*)alloc((size_t)262144 * 2);   // f16 packed weights
  float* dist  = (float*)alloc((size_t)B * N0 * 4);
  int*   fidx  = (int*)  alloc((size_t)B * 1024 * 4);
  int*   gidx  = (int*)  alloc((size_t)B * 1024 * 32 * 4);
  int*   knn_i = (int*)  alloc((size_t)B * N0 * 3 * 4);
  float* knn_w = (float*)alloc((size_t)B * N0 * 3 * 4);
  float* ssum  = (float*)alloc(512 * 4);
  float* ssq   = (float*)alloc(512 * 4);
  float* l1x = (float*)alloc((size_t)B * 3 * 1024 * 4);
  float* l2x = (float*)alloc((size_t)B * 3 * 256 * 4);
  float* l3x = (float*)alloc((size_t)B * 3 * 64 * 4);
  float* l4x = (float*)alloc((size_t)B * 3 * 16 * 4);
  float* l1p = (float*)alloc((size_t)B * 64 * 1024 * 4);
  float* l2p = (float*)alloc((size_t)B * 128 * 256 * 4);
  float* l3p = (float*)alloc((size_t)B * 256 * 64 * 4);
  float* l4p = (float*)alloc((size_t)B * 512 * 16 * 4);
  float* l3f = (float*)alloc((size_t)B * 256 * 64 * 4);
  float* l2f = (float*)alloc((size_t)B * 256 * 256 * 4);
  float* l1f = (float*)alloc((size_t)B * 128 * 1024 * 4);

  // ---- MLP stack: prepack W -> GEMM(+bias+stats) -> BN+ReLU (f32 + f16) ----
  auto run_mlp = [&](_Float16* x16, int M, const Lyr* ls, int nl) -> float* {
    for (int i = 0; i < nl; ++i) {
      const int Kp = r32(ls[i].ci), Cop = r64(ls[i].co), co = ls[i].co;
      k_prepack_w<<<capgrid(Cop * Kp), 256, 0, stream>>>(ls[i].W, Wh, ls[i].ci, co, Kp, Cop);
      k_zero_stats<<<2, 256, 0, stream>>>(ssum, ssq, co);
      dim3 grid(M / BM, Cop / BN);
      k_gemm_wmma<<<grid, 256, 0, stream>>>(x16, Wh, ls[i].b, bufY, ssum, ssq, M, Kp, co);
      _Float16* xn = (i + 1 < nl) ? ((x16 == bufH0) ? bufH1 : bufH0) : (_Float16*)nullptr;
      k_bnrelu<<<capgrid(M * co), 256, 0, stream>>>(bufY, xn, M * co, co, ssum, ssq,
                                                    ls[i].g, ls[i].bt, 1.f / (float)M);
      if (xn) x16 = xn;
    }
    return bufY;
  };

  // ---- Set abstraction ----
  auto run_sa = [&](const float* xyzIn, int N, int xbs, const float* ptsIn, int C,
                    int S, float radius, const Lyr* ls, int nl,
                    float* outXyz, float* outPts) {
    k_fps<<<B, 1024, 0, stream>>>(xyzIn, N, xbs, dist, fidx, S);
    int bs = B * S;
    k_gather_xyz<<<cdiv(bs, 256), 256, 0, stream>>>(xyzIn, N, xbs, fidx, outXyz, B, S);
    k_ball<<<cdiv(bs, 256), 256, 0, stream>>>(xyzIn, N, xbs, outXyz, gidx, B, S,
                                              radius * radius);
    int Cg = 3 + C, Kp = r32(Cg), M = B * S * 32;
    k_group<<<capgrid(M * Kp), 256, 0, stream>>>(xyzIn, N, xbs, ptsIn, C, outXyz,
                                                 gidx, bufH0, B, S, Cg, Kp);
    float* y = run_mlp(bufH0, M, ls, nl);
    int co = ls[nl - 1].co;
    k_maxpool<<<cdiv(B * S * co, 256), 256, 0, stream>>>(y, outPts, B, S, co);
  };

  // ---- Feature propagation ----
  auto run_fp = [&](const float* x1, int N1, int b1s, const float* x2, int N2,
                    const float* pts1, int C1, const float* pts2, int C2,
                    const Lyr* ls, int nl, float* outPts) {
    int t = B * N1;
    k_knn3<<<cdiv(t, 256), 256, 0, stream>>>(x1, N1, b1s, x2, N2, 3 * N2,
                                             knn_i, knn_w, B);
    int Kp = r32(C1 + C2);
    k_interp<<<capgrid(t * Kp), 256, 0, stream>>>(pts1, C1, pts2, C2, N1, N2,
                                                  knn_i, knn_w, bufH0, B, Kp);
    float* y = run_mlp(bufH0, B * N1, ls, nl);
    int co = ls[nl - 1].co;
    k_rows2chw<<<capgrid(B * N1 * co), 256, 0, stream>>>(y, outPts, B, N1, co);
  };

  // ---- pipeline ----
  run_sa(xyz, 4096, 6 * 4096, xyz,  6, 1024, 0.1f, sa1, 3, l1x, l1p);
  run_sa(l1x, 1024, 3 * 1024, l1p, 64,  256, 0.2f, sa2, 3, l2x, l2p);
  run_sa(l2x,  256, 3 *  256, l2p, 128,  64, 0.4f, sa3, 3, l3x, l3p);
  run_sa(l3x,   64, 3 *   64, l3p, 256,  16, 0.8f, sa4, 3, l4x, l4p);

  run_fp(l3x,   64, 3 *   64, l4x,   16, l3p, 256, l4p, 512, fp4, 2, l3f);
  run_fp(l2x,  256, 3 *  256, l3x,   64, l2p, 128, l3f, 256, fp3, 2, l2f);
  run_fp(l1x, 1024, 3 * 1024, l2x,  256, l1p,  64, l2f, 256, fp2, 2, l1f);
  run_fp(xyz, 4096, 6 * 4096, l1x, 1024, nullptr, 0, l1f, 128, fp1, 3, (float*)d_out);
}
